// GlobalLocalLabel_59236188946660
// MI455X (gfx1250) — compile-verified, matrix operand order checked by reading, not run on hardware
//
#include <hip/hip_runtime.h>

typedef __attribute__((ext_vector_type(16))) _Float16 v16h;
typedef __attribute__((ext_vector_type(8)))  _Float16 v8h;
typedef __attribute__((ext_vector_type(8)))  float    v8f;
typedef __attribute__((ext_vector_type(4)))  unsigned int u32x4;
typedef __attribute__((ext_vector_type(8)))  unsigned int u32x8;

#define C_DIM   256
#define K_DIM   2048
#define HW_DIM  3136
#define NTILE   128            // N columns per iteration (8 waves * 16)
#define NITER   (K_DIM / NTILE)
#define TOPK    16

// LDS carve (bytes), base of dynamic LDS assumed 0 (no static __shared__ here):
//   As   @ 0      : 16  x 256  f16 =   8192
//   Bst0 @ 8192   : 128 x 256  f16 =  65536
//   Bst1 @ 73728  : 128 x 256  f16 =  65536
//   Sim  @ 139264 : 16  x 2048 f32 = 131072
#define LDS_AS    0
#define LDS_BST0  8192
#define LDS_BST1  (8192 + 65536)
#define LDS_SIM   (8192 + 2 * 65536)
#define LDS_BYTES (8192 + 2 * 65536 + 131072)

#define TILE_ELEMS (NTILE * C_DIM)   // 32768 f16 elements = 64KB per B tile

__global__ void zero_f32_kernel(float* __restrict__ p, int n) {
    int i = blockIdx.x * 256 + threadIdx.x;
    if (i < n) p[i] = 0.0f;
}

// centersT[n][c] = (f16) centers[c][n]  -- n-major so each block's B tile is contiguous
__global__ void convert_centers_kernel(const float* __restrict__ c,
                                       _Float16* __restrict__ ct) {
    int flat = blockIdx.x * 256 + threadIdx.x;       // 2048*256 total
    int n  = flat & (K_DIM - 1);                     // fastest -> coalesced read
    int cc = flat >> 11;
    ct[(size_t)n * C_DIM + cc] = (_Float16)c[(size_t)cc * K_DIM + n];
}

// One TDM transfer: 64KB contiguous global -> LDS (2D descriptor, 1 row of 32768 B16)
__device__ __forceinline__ void tdm_issue(unsigned lds_addr, const _Float16* gptr) {
    unsigned long long ga = (unsigned long long)gptr;
    u32x4 g0;
    g0.x = 1u;                                            // count=1, user mode
    g0.y = lds_addr;                                      // lds_addr [63:32]
    g0.z = (unsigned)(ga & 0xFFFFFFFFull);                // global_addr lo
    g0.w = (unsigned)((ga >> 32) & 0x01FFFFFFull)         // global_addr hi (57b)
         | (2u << 30);                                    // type=2 ("image")
    u32x8 g1;
    g1.s0 = 0x00010000u;                                  // wg_mask=0, data_size=1 (2B)
    g1.s1 = (TILE_ELEMS & 0xFFFFu) << 16;                 // tensor_dim0 lo16 @ [63:48]
    g1.s2 = (TILE_ELEMS >> 16) | (1u << 16);              // tensor_dim0 hi16 | tensor_dim1=1
    g1.s3 = ((unsigned)TILE_ELEMS) << 16;                 // tile_dim0 = 32768 @ [127:112]
    g1.s4 = 1u;                                           // tile_dim1=1, tile_dim2=0
    g1.s5 = (unsigned)TILE_ELEMS;                         // tensor_dim0_stride lo32
    g1.s6 = 0u;
    g1.s7 = 0u;
    u32x4 gz = {0u, 0u, 0u, 0u};                          // groups 2/3: dims/tiles = 0
    asm volatile("tensor_load_to_lds %0, %1, %2, %3"
                 :: "s"(g0), "s"(g1), "s"(gz), "s"(gz)
                 : "memory");
}

__global__ void __launch_bounds__(256)
glabel_kernel(const float* __restrict__ x,
              const _Float16* __restrict__ centersT,   // [2048][256] f16 (n-major)
              const float* __restrict__ intercept,
              float* __restrict__ gout,   // [16][2048] accumulated via atomics
              float* __restrict__ lout)   // [16*3136] argmax index as float
{
    extern __shared__ char smem[];
    _Float16* As   = (_Float16*)(smem + LDS_AS);     // [16][256]
    _Float16* Bst0 = (_Float16*)(smem + LDS_BST0);   // [128][256] (n-major)
    _Float16* Bst1 = (_Float16*)(smem + LDS_BST1);
    float*    Sim  = (float*)   (smem + LDS_SIM);    // [16][2048]

    const int tid  = threadIdx.x;
    const int lane = tid & 31;
    const int wave = tid >> 5;

    const int mBase  = blockIdx.x * 16;
    const int b      = mBase / HW_DIM;     // tiles never straddle batches (3136 % 16 == 0)
    const int hwBase = mBase % HW_DIM;

    // ---- Kick off TDM prefetch of B tile 0 while we stage A ----
    if (wave == 0) {
        tdm_issue(LDS_BST0, centersT);     // nBase = 0
    }

    // ---- Load A tile: As[r][c] = x[b][c][hwBase + r], coalesced along hw ----
    {
        const int r  = tid & 15;
        const int c0 = tid >> 4;           // 0..15
        #pragma unroll
        for (int cc = 0; cc < C_DIM; cc += 16) {
            const int c = c0 + cc;
            const float v = x[(size_t)(b * C_DIM + c) * HW_DIM + hwBase + r];
            As[r * C_DIM + c] = (_Float16)v;
        }
    }
    __syncthreads();

    const float icpt = intercept[0];
    const int   hs   = (lane >> 4);        // K-half select per ISA 16-bit layout
    const int   rsel = lane & 15;          // A: M row; B: N column

    // ---- Hoist all 8 A fragments (full C=256) into registers ----
    v16h afrag[8];
    {
        const _Float16* arow = As + rsel * C_DIM;
        #pragma unroll
        for (int kb = 0; kb < 8; ++kb) {
            v8h a0 = *(const v8h*)(arow + kb * 32 +      8 * hs);
            v8h a1 = *(const v8h*)(arow + kb * 32 + 16 + 8 * hs);
            afrag[kb] = __builtin_shufflevector(a0, a1,
                            0,1,2,3,4,5,6,7,8,9,10,11,12,13,14,15);
        }
    }

    int buf = 0;
    for (int it = 0; it < NITER; ++it) {
        // Wait for this iteration's TDM tile, make it visible to all waves.
        if (wave == 0) __builtin_amdgcn_s_wait_tensorcnt(0);
        __syncthreads();   // also fences previous iteration's reads of buf^1

        // Prefetch next tile into the other buffer (overlaps with WMMA below).
        if (wave == 0 && it + 1 < NITER) {
            tdm_issue(buf ? LDS_BST0 : LDS_BST1,
                      centersT + (size_t)(it + 1) * TILE_ELEMS);
        }

        const _Float16* Bst  = buf ? Bst1 : Bst0;
        const _Float16* brow = Bst + (wave * 16 + rsel) * C_DIM;

        // ---- Preload ALL 8 B fragments, then chain the WMMAs.          ----
        // 16 ds_load_b128 in flight before the first wait: loads for
        // fragments 2..7 overlap with the first WMMAs (partial dscnt waits).
        v16h bfrag[8];
        #pragma unroll
        for (int kb = 0; kb < 8; ++kb) {
            v8h b0 = *(const v8h*)(brow + kb * 32 +      8 * hs);
            v8h b1 = *(const v8h*)(brow + kb * 32 + 16 + 8 * hs);
            bfrag[kb] = __builtin_shufflevector(b0, b1,
                            0,1,2,3,4,5,6,7,8,9,10,11,12,13,14,15);
        }

        v8f acc = {};
        #pragma unroll
        for (int kb = 0; kb < 8; ++kb) {
            acc = __builtin_amdgcn_wmma_f32_16x16x32_f16(
                      false, afrag[kb], false, bfrag[kb], (short)0, acc, false, false);
        }

        // ---- sigmoid (v_exp + v_rcp fast path) + scatter into Sim ----
        const int col = it * NTILE + wave * 16 + rsel;
        #pragma unroll
        for (int r = 0; r < 8; ++r) {
            const float t = acc[r] + icpt;
            const float s = __builtin_amdgcn_rcpf(1.0f + __expf(-t));
            Sim[(r + 8 * hs) * K_DIM + col] = s;     // D layout: VGPR r -> M = r + 8*hs
        }
        buf ^= 1;
    }
    __syncthreads();

    // ---- Per-row top-16 + argmax: wave w handles rows w and w+8 ----
    #pragma unroll
    for (int rr = 0; rr < 2; ++rr) {
        const int row = wave + 8 * rr;
        const float* srow = Sim + row * K_DIM;

        float tv[TOPK];
        int   ti[TOPK];
        #pragma unroll
        for (int q = 0; q < TOPK; ++q) { tv[q] = -1e30f; ti[q] = -1; }

        // lane-local sorted top-16 over a strided (conflict-free) column set
        for (int j = 0; j < K_DIM / 32; ++j) {
            const int col = j * 32 + lane;
            float cv = srow[col];
            if (cv > tv[TOPK - 1]) {
                int ci = col;
                #pragma unroll
                for (int q = 0; q < TOPK; ++q) {       // static-index bubble insert
                    if (cv > tv[q]) {
                        float t1 = tv[q]; int t2 = ti[q];
                        tv[q] = cv; ti[q] = ci;
                        cv = t1;   ci = t2;
                    }
                }
            }
        }

        // merge across 32 lanes: 16 rounds of wave-max (tie -> smaller index)
        for (int s = 0; s < TOPK; ++s) {
            float v = tv[0];
            int   i = ti[0];
            #pragma unroll
            for (int m = 16; m >= 1; m >>= 1) {
                const float ov = __shfl_xor(v, m, 32);
                const int   oi = __shfl_xor(i, m, 32);
                if (ov > v || (ov == v && oi < i)) { v = ov; i = oi; }
            }
            if (ti[0] == i) {                 // winner lane pops its head (indices unique)
                #pragma unroll
                for (int q = 0; q < TOPK - 1; ++q) { tv[q] = tv[q + 1]; ti[q] = ti[q + 1]; }
                tv[TOPK - 1] = -1e30f; ti[TOPK - 1] = -1;
            }
            if (lane == 0) {
                atomicAdd(gout + b * K_DIM + i, v * (1.0f / (float)HW_DIM));
                if (s == 0) lout[mBase + row] = (float)i;   // argmax, first occurrence
            }
        }
    }
}

extern "C" void kernel_launch(void* const* d_in, const int* in_sizes, int n_in,
                              void* d_out, int out_size, void* d_ws, size_t ws_size,
                              hipStream_t stream) {
    (void)in_sizes; (void)n_in; (void)ws_size; (void)out_size;

    const float* x         = (const float*)d_in[0];   // [16][256][56][56]
    const float* centers   = (const float*)d_in[1];   // [256][2048]
    const float* intercept = (const float*)d_in[2];   // [1]

    _Float16* centersT = (_Float16*)d_ws;             // [2048][256] f16 (1 MB scratch)

    float* gout = (float*)d_out;                      // [16][2048]
    float* lout = (float*)d_out + 16 * K_DIM;         // [16*3136] as float

    // Pre-pass 1: transpose+convert centers to f16 n-major (feeds contiguous TDM tiles).
    convert_centers_kernel<<<(K_DIM * C_DIM) / 256, 256, 0, stream>>>(centers, centersT);

    // Pre-pass 2: d_out is poisoned; global_label is accumulated atomically -> zero it.
    zero_f32_kernel<<<(16 * K_DIM + 255) / 256, 256, 0, stream>>>(gout, 16 * K_DIM);

    static bool attr_set = false;   // idempotent, deterministic, capture-safe
    if (!attr_set) {
        hipFuncSetAttribute((const void*)glabel_kernel,
                            hipFuncAttributeMaxDynamicSharedMemorySize, LDS_BYTES);
        attr_set = true;
    }

    const int nBlocks = (16 * HW_DIM) / 16;           // 3136
    glabel_kernel<<<nBlocks, 256, LDS_BYTES, stream>>>(x, centersT, intercept, gout, lout);
}
